// Network_26611617366437
// MI455X (gfx1250) — matrix operand order repaired and checked by reading, not compiled, not run on hardware
//
#include <hip/hip_runtime.h>
#include <hip/hip_bf16.h>

typedef __attribute__((ext_vector_type(16))) _Float16 v16h;
typedef __attribute__((ext_vector_type(8)))  _Float16 v8h;
typedef __attribute__((ext_vector_type(8)))  float    v8f;
typedef __attribute__((ext_vector_type(4)))  int      v4i;

#define DIM 128
#define NNB 64
#define ETS 72   // LDS row stride in halves: 144B -> 16B aligned, reduced bank conflicts

// Async global<->LDS path (CDNA5): guarded so we fall back cleanly if the
// toolchain doesn't declare these builtins. Builtin params are v4i pointers
// in addrspace(1) (global) and addrspace(3) (LDS).
#if defined(__HIP_DEVICE_COMPILE__) && \
    __has_builtin(__builtin_amdgcn_global_load_async_to_lds_b128) && \
    __has_builtin(__builtin_amdgcn_global_store_async_from_lds_b128)
#define HAVE_ASYNC_LDS 1
typedef __attribute__((address_space(1))) v4i gv4i;
typedef __attribute__((address_space(3))) v4i lv4i;
__device__ __forceinline__ void wait_asynccnt0() {
#if __has_builtin(__builtin_amdgcn_s_wait_asynccnt)
    __builtin_amdgcn_s_wait_asynccnt(0);
#else
    asm volatile("s_wait_asynccnt 0x0" ::: "memory");
#endif
}
#else
#define HAVE_ASYNC_LDS 0
#endif

__global__ __launch_bounds__(128)
void kge_fused_kernel(const int* __restrict__ data_r,
                      const int* __restrict__ data_e,
                      const int* __restrict__ rel,
                      const int* __restrict__ pos_id,
                      const int* __restrict__ neg_id,
                      const float* __restrict__ ent_emb,
                      const float* __restrict__ edge_w,
                      const float* __restrict__ rel_emb,
                      float* __restrict__ out, int B)
{
    // Transposed f16 neighbor embeddings: Et[d][n], d in [0,128), n in [0,64)
    __shared__ __align__(16) _Float16 Et[DIM * ETS];
    __shared__ float sLog[NNB];
    __shared__ float sW[NNB];
    __shared__ int   sIdx[NNB];
#if HAVE_ASYNC_LDS
    __shared__ __align__(16) float sCopy[2 * DIM];   // staging for pos/neg rows
#endif

    const int b    = blockIdx.x;
    const int t    = threadIdx.x;
    const int lane = t & 31;          // wave32
    const int wave = t >> 5;          // 4 waves per block
    const int hi   = lane >> 4;       // lane half (0/1)
    const int nl   = lane & 15;       // column within WMMA tile

#if HAVE_ASYNC_LDS
    // Kick off the pos/neg row copies early: DMA global -> LDS on ASYNCcnt,
    // overlapping all of the softmax + gather work below. 16B/lane x 32 lanes
    // = one full 512B embedding row per wave. Wave0 = pos, wave1 = neg.
    if (wave < 2) {
        const int id = (wave == 0) ? pos_id[b] : neg_id[b];
        const float* src = ent_emb + (size_t)id * DIM + lane * 4;
        float* dstL = &sCopy[wave * DIM + lane * 4];
        __builtin_amdgcn_global_load_async_to_lds_b128(
            (gv4i*)src, (lv4i*)dstL, 0, 0);
    }
#endif

    // ---- Phase 1: gather edge-weight logits + neighbor indices ----
    if (t < NNB) {
        sIdx[t] = data_e[(size_t)b * NNB + t];
        sLog[t] = edge_w[data_r[(size_t)b * NNB + t]];
    }
    __syncthreads();

    // Softmax over 64 neighbors (redundant per-thread; LDS broadcast reads)
    float mx = -3.4e38f;
#pragma unroll 8
    for (int n = 0; n < NNB; ++n) mx = fmaxf(mx, sLog[n]);
    float sm = 0.f;
#pragma unroll 8
    for (int n = 0; n < NNB; ++n) sm += __expf(sLog[n] - mx);
    if (t < NNB) sW[t] = __expf(sLog[t] - mx) / sm;

    // ---- Phase 2: gather neighbor embedding rows (coalesced 512B per row),
    //      store transposed to LDS as f16 ----
#pragma unroll 4
    for (int n = 0; n < NNB; ++n) {
        if (n + 4 < NNB)
            __builtin_prefetch(&ent_emb[(size_t)sIdx[n + 4] * DIM + t], 0, 0);
        Et[(size_t)t * ETS + n] = (_Float16)ent_emb[(size_t)sIdx[n] * DIM + t];
    }

#if HAVE_ASYNC_LDS
    // Drain the async loads, then DMA LDS -> out (pos/neg rows), again with
    // no VGPR round-trip. Kernel-end implicit wait-idle covers completion.
    if (wave < 2) {
        wait_asynccnt0();
        float* dstG = out + (size_t)(wave + 1) * B * DIM + (size_t)b * DIM + lane * 4;
        const float* srcL = &sCopy[wave * DIM + lane * 4];
        __builtin_amdgcn_global_store_async_from_lds_b128(
            (gv4i*)dstG, (lv4i*)srcL, 0, 0);
    }
#else
    // Fallback: plain coalesced per-thread copies
    out[(size_t)B * DIM + (size_t)b * DIM + t]     = ent_emb[(size_t)pos_id[b] * DIM + t];
    out[(size_t)2 * B * DIM + (size_t)b * DIM + t] = ent_emb[(size_t)neg_id[b] * DIM + t];
#endif

    __syncthreads();

    // ---- Phase 3: Eh = w (1x64) x E (64x128) via v_wmma_f32_16x16x32_f16 ----
    // A (16x32 f16) layout: lane<16 holds M=lane; element e -> K = e + hi*16.
    // Broadcast the same w into every M row so all D rows are identical.
    union U16 { v16h v; _Float16 h[16]; };
    U16 a0, a1;                       // K in [0,32) and [32,64)
#pragma unroll
    for (int e = 0; e < 16; ++e) {
        const int kl = e + hi * 16;
        a0.h[e] = (_Float16)sW[kl];
        a1.h[e] = (_Float16)sW[kl + 32];
    }

    const int rv = rel[b];

    // Each wave owns d-tiles {2*wave, 2*wave+1} -> covers d in [32*wave, 32*wave+32)
#pragma unroll
    for (int j = 0; j < 2; ++j) {
        const int dt    = 2 * wave + j;
        const int dbase = dt * 16;
        // B (32x16 f16) layout: lane<16 -> col N=lane, elements e -> K=e (+16 for hi half)
        const _Float16* brow = &Et[(size_t)(dbase + nl) * ETS + hi * 16];

        v8f c = {0.f, 0.f, 0.f, 0.f, 0.f, 0.f, 0.f, 0.f};

        v8h b0lo = *(const v8h*)(brow);
        v8h b0hi = *(const v8h*)(brow + 8);
        v16h b0  = __builtin_shufflevector(b0lo, b0hi,
                     0,1,2,3,4,5,6,7,8,9,10,11,12,13,14,15);
        c = __builtin_amdgcn_wmma_f32_16x16x32_f16(false, a0.v, false, b0,
                                                   (short)0, c, false, false);

        v8h b1lo = *(const v8h*)(brow + 32);
        v8h b1hi = *(const v8h*)(brow + 40);
        v16h b1  = __builtin_shufflevector(b1lo, b1hi,
                     0,1,2,3,4,5,6,7,8,9,10,11,12,13,14,15);
        c = __builtin_amdgcn_wmma_f32_16x16x32_f16(false, a1.v, false, b1,
                                                   (short)0, c, false, false);

        // All D rows identical -> c[0] holds Eh[dbase + (lane&15)]
        if (lane < 16) {
            const int d = dbase + lane;
            out[(size_t)b * DIM + d] = c[0] + rel_emb[(size_t)rv * DIM + d];
        }
    }
}

extern "C" void kernel_launch(void* const* d_in, const int* in_sizes, int n_in,
                              void* d_out, int out_size, void* d_ws, size_t ws_size,
                              hipStream_t stream) {
    const int*   data_r  = (const int*)d_in[0];
    const int*   data_e  = (const int*)d_in[1];
    const int*   rel     = (const int*)d_in[2];
    const int*   pos_id  = (const int*)d_in[3];
    const int*   neg_id  = (const int*)d_in[4];
    const float* ent_emb = (const float*)d_in[5];
    const float* edge_w  = (const float*)d_in[6];
    const float* rel_emb = (const float*)d_in[7];
    float* out = (float*)d_out;

    const int B = in_sizes[2];  // rel has shape [B]

    kge_fused_kernel<<<B, 128, 0, stream>>>(data_r, data_e, rel, pos_id, neg_id,
                                            ent_emb, edge_w, rel_emb, out, B);
}
